// WindowAttentionDISTILL_6012954214726
// MI455X (gfx1250) — matrix operand order repaired and checked by reading, not compiled, not run on hardware
//
#include <hip/hip_runtime.h>

typedef __attribute__((ext_vector_type(16))) _Float16 v16h;
typedef __attribute__((ext_vector_type(8)))  float    v8f;
typedef _Float16 half_t;

// ---------------------------------------------------------------------------
// Problem constants
// ---------------------------------------------------------------------------
#define WIN_N   49                 // tokens per window
#define CH      384
#define NHEAD   12
#define HD      32
#define NBATCH  4096
#define MROWS   (NBATCH * WIN_N)   // 200704 = 1568 * 128
#define NWMASK  64

// Async global->LDS copy support (gfx1250); guarded so either path compiles.
#if defined(__has_builtin)
#if __has_builtin(__builtin_amdgcn_global_load_async_to_lds_b128) && \
    __has_builtin(__builtin_amdgcn_s_wait_asynccnt)
#define HAVE_ASYNC_LDS 1
#endif
#endif
#ifndef HAVE_ASYNC_LDS
#define HAVE_ASYNC_LDS 0
#endif

#if HAVE_ASYNC_LDS
typedef int v4i __attribute__((vector_size(16)));
typedef v4i __attribute__((address_space(1))) *g4p;   // global 16B-vector ptr
typedef v4i __attribute__((address_space(3))) *l4p;   // LDS    16B-vector ptr
#endif

// ---------------------------------------------------------------------------
// WMMA fragment loaders (contiguous k order, same for A and B, wave32)
// A 16x32 (MxK): lane&15 = M; VGPR v: k0 = 2v + 16*(lane>>4)
//   -> halves 0..7 / 8..15 contiguous per lane: 2x ds_load_b128
// B 32x16 (KxN): lane&15 = N, (lane>>4)*16 = K group, VGPR v = pair 2v
//   -> 16 contiguous halves per lane: 2x ds_load_b128
// Rows stored K-contiguous with stride `ld` halves (40 -> 80B, 16B aligned).
// ---------------------------------------------------------------------------
__device__ inline v16h frag_a_load(const half_t* tile, int ld) {
  const int lane = threadIdx.x & 31;
  const int g = lane >> 4;
  const half_t* p = tile + (lane & 15) * ld;
  v16h f;
#pragma unroll
  for (int v = 0; v < 8; ++v) {
    const int k0 = 2 * v + 16 * g;
    f[2 * v]     = p[k0];
    f[2 * v + 1] = p[k0 + 1];
  }
  return f;
}

__device__ inline v16h frag_b_load(const half_t* tile, int ld) {
  const int lane = threadIdx.x & 31;
  const half_t* p = tile + (lane & 15) * ld + (lane >> 4) * 16;
  v16h f;
#pragma unroll
  for (int v = 0; v < 8; ++v) {
    f[2 * v]     = p[2 * v];
    f[2 * v + 1] = p[2 * v + 1];
  }
  return f;
}

// ---------------------------------------------------------------------------
// 128x128x32 tiled WMMA GEMM with LDS double buffering:
//   out[M][Ncols] = A[M][K] * W[Ncols][K]^T + bias
// 256 threads = 8 waves; wave computes 32x64 (2x4 tiles of 16x16).
// Global loads for tile k+1 overlap WMMA on tile k; prefetch tile k+2.
// EPI==0 : A fp32 (x), writes q/k/v fp32 slices of d_out + f16 qkv to ws
// EPI==1 : A f16 (ctx), writes fp32 `out` (offset 0 of d_out)
// ---------------------------------------------------------------------------
template <int EPI>
__global__ __launch_bounds__(256) void gemm128(const void* __restrict__ Aptr,
                                               const float* __restrict__ W,
                                               const float* __restrict__ bias,
                                               float* __restrict__ outf,
                                               half_t* __restrict__ outh,
                                               int M, int K) {
  constexpr int BK  = 32;
  constexpr int LDT = 40;                  // 80B row stride: aligned, low-conflict
  __shared__ half_t As[2][128 * LDT];
  __shared__ half_t Bs[2][128 * LDT];

  const int tid  = threadIdx.x;
  const int lane = tid & 31;
  const int wave = tid >> 5;
  const int bm = blockIdx.x * 128;
  const int bn = blockIdx.y * 128;
  const int wm = (wave & 3) * 32;
  const int wn = (wave >> 2) * 64;

  v8f acc[2][4];
#pragma unroll
  for (int i = 0; i < 2; ++i)
#pragma unroll
    for (int j = 0; j < 4; ++j) {
      v8f z = {0.f, 0.f, 0.f, 0.f, 0.f, 0.f, 0.f, 0.f};
      acc[i][j] = z;
    }

  const int lrow = tid >> 1;               // 0..127
  const int lkc  = (tid & 1) * 16;         // 0 / 16

  half_t ra[16], rb[16];

  auto load_regs = [&](int kb) {
    if (EPI == 0) {
      const float* s = (const float*)Aptr + (size_t)(bm + lrow) * K + kb + lkc;
#pragma unroll
      for (int i = 0; i < 16; ++i) ra[i] = (half_t)s[i];
    } else {
      const half_t* s = (const half_t*)Aptr + (size_t)(bm + lrow) * K + kb + lkc;
#pragma unroll
      for (int i = 0; i < 16; ++i) ra[i] = s[i];
    }
    const float* w = W + (size_t)(bn + lrow) * K + kb + lkc;
#pragma unroll
    for (int i = 0; i < 16; ++i) rb[i] = (half_t)w[i];
  };
  auto store_regs = [&](int buf) {
#pragma unroll
    for (int i = 0; i < 16; ++i) As[buf][lrow * LDT + lkc + i] = ra[i];
#pragma unroll
    for (int i = 0; i < 16; ++i) Bs[buf][lrow * LDT + lkc + i] = rb[i];
  };

  load_regs(0);
  store_regs(0);
  __syncthreads();

  int cur = 0;
  for (int kb = 0; kb < K; kb += BK) {
    const bool last = (kb + BK >= K);
    if (!last) {
      load_regs(kb + BK);                  // overlaps with WMMA below
      if (kb + 2 * BK < K) {               // prefetch tile k+2 into caches
        if (EPI == 0)
          __builtin_prefetch((const float*)Aptr + (size_t)(bm + lrow) * K + kb + 2 * BK + lkc, 0, 0);
        else
          __builtin_prefetch((const half_t*)Aptr + (size_t)(bm + lrow) * K + kb + 2 * BK + lkc, 0, 0);
        __builtin_prefetch(W + (size_t)(bn + lrow) * K + kb + 2 * BK + lkc, 0, 0);
      }
    }
#pragma unroll
    for (int i = 0; i < 2; ++i) {
      v16h a = frag_a_load(&As[cur][(wm + 16 * i) * LDT], LDT);
#pragma unroll
      for (int j = 0; j < 4; ++j) {
        v16h b = frag_b_load(&Bs[cur][(wn + 16 * j) * LDT], LDT);
        acc[i][j] = __builtin_amdgcn_wmma_f32_16x16x32_f16(
            false, a, false, b, (short)0, acc[i][j], false, false);
      }
    }
    if (!last) {
      store_regs(cur ^ 1);
      __syncthreads();
      cur ^= 1;
    }
  }

  // epilogue
  const int g = lane >> 4;
#pragma unroll
  for (int i = 0; i < 2; ++i) {
#pragma unroll
    for (int j = 0; j < 4; ++j) {
      const int c = bn + wn + 16 * j + (lane & 15);
      const float bj = bias[c];
#pragma unroll
      for (int v = 0; v < 8; ++v) {
        const int r = bm + wm + 16 * i + v + 8 * g;
        const float val = acc[i][j][v] + bj;
        if (EPI == 0) {
          const int which = c / CH;        // 0=q,1=k,2=v
          const int cc = c - which * CH;
          outf[(size_t)(1 + which) * M * CH + (size_t)r * CH + cc] = val;
          outh[(size_t)r * (3 * CH) + c] = (half_t)val;
        } else {
          outf[(size_t)r * CH + c] = val;
        }
      }
    }
  }
}

// ---------------------------------------------------------------------------
// Fused per-window attention: one block per window, 12 waves = 12 heads.
//   S_h = scale * q_h k_h^T           (WMMA f16, padded 49->64)
//   t_g = proj_l-mix(S) + rpb + mask  -> softmax rows (shfl_xor, wave32)
//   ctx_g = (proj_w-mix(soft) + b) @ v_g   (VALU, lane = channel, hd == 32)
// Dynamic LDS: q/k/v [12][64][40] f16 + S [12][49][52] fp32 = 306,624 B.
// q/k/v staged with GLOBAL_LOAD_ASYNC_TO_LDS_B128 when available.
// ---------------------------------------------------------------------------
#define QLD   40
#define QT    (12 * 64 * QLD)
#define S_LD  52
#define ATTN_SMEM (3 * QT * 2 + 12 * WIN_N * S_LD * 4)   // 306624 bytes

__global__ __launch_bounds__(384) void attn_win(
    const half_t* __restrict__ qkv,   // [M][1152] f16
    const float*  __restrict__ mask,  // [64][49][49]
    const float*  __restrict__ rpb,   // [169][12]
    const int*    __restrict__ rel,   // [49][49]
    const float*  __restrict__ pl_w, const float* __restrict__ pl_b,
    const float*  __restrict__ pw_w, const float* __restrict__ pw_b,
    half_t* __restrict__ ctx)         // [M][384] f16
{
  extern __shared__ char smem[];
  half_t* q_s = (half_t*)smem;
  half_t* k_s = q_s + QT;
  half_t* v_s = k_s + QT;
  float*  S   = (float*)(v_s + QT);   // [12][49][S_LD]

  const int b    = blockIdx.x;
  const int tid  = threadIdx.x;
  const int lane = tid & 31;
  const int wv   = tid >> 5;          // head id 0..11
  const int wi   = b & (NWMASK - 1);  // mask window index = b % nW

  const half_t* base = qkv + (size_t)b * WIN_N * (3 * CH);

  // ---- phase 0: stage q/k/v into LDS, pad rows 49..63 with zeros ----------
  // zero padding rows for q,k (WMMA reads the full 64-row tiles)
  for (int idx = tid; idx < 12 * 15 * 32; idx += 384) {
    const int h = idx / (15 * 32);
    const int r = idx % (15 * 32);
    const int n = WIN_N + (r >> 5);
    const int d = r & 31;
    const int o = (h * 64 + n) * QLD + d;
    q_s[o] = (half_t)0.f;
    k_s[o] = (half_t)0.f;
  }
#if HAVE_ASYNC_LDS
  {
    // 16-byte chunks: [head][token][d-chunk of 8 halves], 2352 per tensor
    for (int ci = tid; ci < 12 * WIN_N * 4; ci += 384) {
      const int h  = ci / (WIN_N * 4);
      const int r  = ci % (WIN_N * 4);
      const int n  = r >> 2;
      const int dc = (r & 3) * 8;
      const half_t* g = base + (size_t)n * (3 * CH) + h * HD + dc;
      const int o = (h * 64 + n) * QLD + dc;
      __builtin_amdgcn_global_load_async_to_lds_b128(
          (g4p)(g),          (l4p)(q_s + o), 0, 0);
      __builtin_amdgcn_global_load_async_to_lds_b128(
          (g4p)(g + CH),     (l4p)(k_s + o), 0, 0);
      __builtin_amdgcn_global_load_async_to_lds_b128(
          (g4p)(g + 2 * CH), (l4p)(v_s + o), 0, 0);
    }
    __builtin_amdgcn_s_wait_asynccnt(0);
  }
#else
  for (int idx = tid; idx < 12 * WIN_N * 32; idx += 384) {
    const int h = idx / (WIN_N * 32);
    const int r = idx % (WIN_N * 32);
    const int n = r >> 5;
    const int d = r & 31;
    const half_t* p = base + (size_t)n * (3 * CH) + h * HD + d;
    const int o = (h * 64 + n) * QLD + d;
    q_s[o] = p[0];
    k_s[o] = p[CH];
    v_s[o] = p[2 * CH];
  }
#endif
  __syncthreads();

  // ---- phase 1: S = scale * q k^T per head (wave-per-head WMMA) -----------
  const float scale = 0.17677669529663687f;   // 32^-0.5
  const half_t* qh = q_s + wv * 64 * QLD;
  const half_t* kh = k_s + wv * 64 * QLD;
#pragma unroll
  for (int ti = 0; ti < 4; ++ti) {
    v16h a = frag_a_load(qh + (ti * 16) * QLD, QLD);
#pragma unroll
    for (int tj = 0; tj < 4; ++tj) {
      v16h bf = frag_b_load(kh + (tj * 16) * QLD, QLD);
      v8f d = {0.f, 0.f, 0.f, 0.f, 0.f, 0.f, 0.f, 0.f};
      d = __builtin_amdgcn_wmma_f32_16x16x32_f16(false, a, false, bf,
                                                 (short)0, d, false, false);
      const int cc = tj * 16 + (lane & 15);
      if (cc < WIN_N) {
        const int rbase = ti * 16 + 8 * (lane >> 4);
#pragma unroll
        for (int v = 0; v < 8; ++v) {
          const int rr = rbase + v;
          if (rr < WIN_N) S[(wv * WIN_N + rr) * S_LD + cc] = scale * d[v];
        }
      }
    }
  }
  __syncthreads();

  // ---- phase 2: proj_l head-mix + bias + mask, softmax, rewrite S ---------
  float pl[12], pw[12];
#pragma unroll
  for (int h = 0; h < 12; ++h) { pl[h] = pl_w[wv * 12 + h]; pw[h] = pw_w[wv * 12 + h]; }
  const float plb = pl_b[wv];
  const float pwb = pw_b[wv];

  for (int nc = 0; nc < WIN_N; nc += 7) {     // 49 = 7 chunks of 7 rows
    float t[7][2];
#pragma unroll
    for (int r = 0; r < 7; ++r) {
      const int n = nc + r;
      float rowmax = -3.0e38f;
#pragma unroll
      for (int p = 0; p < 2; ++p) {
        const int m = lane + 32 * p;
        float val = -3.0e38f;
        if (m < WIN_N) {
          float a2 = plb + mask[(wi * WIN_N + n) * WIN_N + m]
                         + rpb[rel[n * WIN_N + m] * NHEAD + wv];
#pragma unroll
          for (int h = 0; h < 12; ++h) a2 += pl[h] * S[(h * WIN_N + n) * S_LD + m];
          val = a2;
        }
        t[r][p] = val;
        rowmax = fmaxf(rowmax, val);
      }
#pragma unroll
      for (int off = 16; off > 0; off >>= 1)
        rowmax = fmaxf(rowmax, __shfl_xor(rowmax, off, 32));
      float sum = 0.f;
#pragma unroll
      for (int p = 0; p < 2; ++p) {
        const float e = (t[r][p] > -1.0e37f) ? __expf(t[r][p] - rowmax) : 0.f;
        t[r][p] = e;
        sum += e;
      }
#pragma unroll
      for (int off = 16; off > 0; off >>= 1) sum += __shfl_xor(sum, off, 32);
      const float inv = 1.0f / sum;
      t[r][0] *= inv;
      t[r][1] *= inv;
    }
    __syncthreads();   // all heads finished reading this row chunk
#pragma unroll
    for (int r = 0; r < 7; ++r) {
      const int n = nc + r;
#pragma unroll
      for (int p = 0; p < 2; ++p) {
        const int m = lane + 32 * p;
        if (m < WIN_N) S[(wv * WIN_N + n) * S_LD + m] = t[r][p];
      }
    }
    __syncthreads();
  }

  // ---- phase 3: ctx_g = (proj_w-mix(soft) + b) @ v_g ; lane = channel -----
  const half_t* vh = v_s + wv * 64 * QLD;
  half_t* cb = ctx + (size_t)b * WIN_N * CH + wv * HD + lane;
  for (int n = 0; n < WIN_N; ++n) {
    float acc3 = 0.f;
    for (int m = 0; m < WIN_N; ++m) {
      float pmix = pwb;
#pragma unroll
      for (int h = 0; h < 12; ++h) pmix += pw[h] * S[(h * WIN_N + n) * S_LD + m];
      acc3 += pmix * (float)vh[m * QLD + lane];
    }
    cb[(size_t)n * CH] = (half_t)acc3;
  }
}

// ---------------------------------------------------------------------------
// Host launcher
// ---------------------------------------------------------------------------
extern "C" void kernel_launch(void* const* d_in, const int* in_sizes, int n_in,
                              void* d_out, int out_size, void* d_ws, size_t ws_size,
                              hipStream_t stream) {
  (void)in_sizes; (void)n_in; (void)out_size; (void)ws_size;
  const float* x     = (const float*)d_in[0];
  const float* mask  = (const float*)d_in[1];
  const float* qkv_w = (const float*)d_in[2];
  const float* qkv_b = (const float*)d_in[3];
  const float* rpb   = (const float*)d_in[4];
  const float* pl_w  = (const float*)d_in[5];
  const float* pl_b  = (const float*)d_in[6];
  const float* pw_w  = (const float*)d_in[7];
  const float* pw_b  = (const float*)d_in[8];
  const float* pjw   = (const float*)d_in[9];
  const float* pjb   = (const float*)d_in[10];
  const int*   rel   = (const int*)d_in[11];

  float*  out    = (float*)d_out;
  half_t* ws_qkv = (half_t*)d_ws;                       // [M][1152] f16
  half_t* ws_ctx = ws_qkv + (size_t)MROWS * (3 * CH);   // [M][384] f16

  // 1) qkv = x @ qkv_w^T + b -> q/k/v fp32 outputs + f16 workspace copy
  gemm128<0><<<dim3(MROWS / 128, (3 * CH) / 128), 256, 0, stream>>>(
      x, qkv_w, qkv_b, out, ws_qkv, MROWS, CH);

  // 2) fused window attention (per-window block, wave-per-head WMMA)
  attn_win<<<NBATCH, 384, ATTN_SMEM, stream>>>(
      ws_qkv, mask, rpb, rel, pl_w, pl_b, pw_w, pw_b, ws_ctx);

  // 3) out = ctx @ proj_weight^T + proj_bias
  gemm128<1><<<dim3(MROWS / 128, CH / 128), 256, 0, stream>>>(
      ws_ctx, pjw, pjb, out, nullptr, MROWS, CH);
}
